// GraphBuilder_35407710388429
// MI455X (gfx1250) — matrix operand order repaired and checked by reading, not compiled, not run on hardware
//
#include <hip/hip_runtime.h>

#define BQ 2048        // batch (queries)
#define QN 131072      // queue size (features)
#define DD 128         // dim
#define KK 16          // top-k
#define CHUNK 512      // features per inner chunk
#define CAP 64         // candidate-list capacity per query per chunk
#define LISTMAX (16 + CAP)

typedef float  v2f  __attribute__((ext_vector_type(2)));
typedef float  v8f  __attribute__((ext_vector_type(8)));
typedef __bf16 v8bf  __attribute__((ext_vector_type(8)));
typedef __bf16 v16bf __attribute__((ext_vector_type(16)));

// ---------------------------------------------------------------------------
// Split fp32 -> bf16 (hi) + bf16 (residual lo). Memory-bound one-time pass.
// ---------------------------------------------------------------------------
__global__ __launch_bounds__(256) void split_bf16(
    const float* __restrict__ src, __bf16* __restrict__ hi,
    __bf16* __restrict__ lo, int n)
{
  const int base = (blockIdx.x * 256 + threadIdx.x) * 4;
  if (base + 3 < n) {
#pragma unroll
    for (int j = 0; j < 4; ++j) {
      const float x = src[base + j];
      const __bf16 h = (__bf16)x;          // RNE convert
      const float  r = x - (float)h;       // residual
      hi[base + j] = h;
      lo[base + j] = (__bf16)r;
    }
  }
}

// ---------------------------------------------------------------------------
// Phase 1 (fast path): split-bf16 WMMA GEMM + running top-16 per query.
// sim = Ahi*Bhi + Alo*Bhi + Ahi*Blo  (f32 accumulate), rel. err ~2^-16.
// grid = (BQ/16 query tiles, parts), block = 256 threads (8 wave32)
// ---------------------------------------------------------------------------
__global__ __launch_bounds__(256) void sim_topk_part_bf16(
    const __bf16* __restrict__ qhi, const __bf16* __restrict__ qlo,
    const __bf16* __restrict__ fhi, const __bf16* __restrict__ flo,
    float* __restrict__ wsv, int* __restrict__ wsi, int parts, int qpp)
{
  __shared__ float S[16 * CHUNK];       // 32 KB  sim tile
  __shared__ float lv[16][LISTMAX];
  __shared__ int   li[16][LISTMAX];
  __shared__ int   cnt[16];
  __shared__ float nrv[16][16];
  __shared__ int   nri[16][16];
  __shared__ float tau[16];

  const int tid  = threadIdx.x;
  const int lane = tid & 31;
  const int wave = tid >> 5;
  const int half = lane >> 4;
  const int l16  = lane & 15;
  const int qrp  = tid >> 4;
  const int t    = tid & 15;

  const int qt       = blockIdx.x;
  const int part     = blockIdx.y;
  const int partBase = part * qpp;
  const int nchunks  = qpp / CHUNK;
  const float NEGINF = -__builtin_inff();

  if (t == 0) tau[qrp] = NEGINF;

  // A operand preload (16-bit A 16x32): per lane 16 consecutive K:
  // 32s + 16*half .. +15 (elements 0-15), matching the B operand order.
  const int qrow = qt * 16 + l16;
  v16bf Ahi[4], Alo[4];
#pragma unroll
  for (int s = 0; s < 4; ++s) {
    const size_t o0 = (size_t)qrow * DD + 32 * s + 16 * half;
    const v8bf h0 = *(const v8bf*)&qhi[o0];
    const v8bf h1 = *(const v8bf*)&qhi[o0 + 8];
    Ahi[s] = __builtin_shufflevector(h0, h1, 0,1,2,3,4,5,6,7,8,9,10,11,12,13,14,15);
    const v8bf g0 = *(const v8bf*)&qlo[o0];
    const v8bf g1 = *(const v8bf*)&qlo[o0 + 8];
    Alo[s] = __builtin_shufflevector(g0, g1, 0,1,2,3,4,5,6,7,8,9,10,11,12,13,14,15);
  }

  float runval = NEGINF;
  int   runidx = 0;

  for (int c = 0; c < nchunks; ++c) {
    __syncthreads();                    // prior selection done with S

    const int gbase = partBase + c * CHUNK;

    // prefetch next chunk of the (L2-resident) hi/lo banks
    if (c + 1 < nchunks) {
      const size_t nb = (size_t)(partBase + (c + 1) * CHUNK + tid * 2) * DD;
      __builtin_prefetch(&fhi[nb], 0, 3);
      __builtin_prefetch(&fhi[nb + DD], 0, 3);
      __builtin_prefetch(&flo[nb], 0, 3);
      __builtin_prefetch(&flo[nb + DD], 0, 3);
    }

    // ---- GEMM: 32 tiles of 16x16 per chunk, 4 per wave, 12 bf16 WMMAs each
#pragma unroll 1
    for (int j = 0; j < 4; ++j) {
      const int T = wave + 8 * j;
      // B operand (32x16 bf16): per lane 16 consecutive K at 16*half.
      const size_t rowoff = (size_t)(gbase + T * 16 + l16) * DD + 16 * half;
      v8f acc = {};
#pragma unroll
      for (int s = 0; s < 4; ++s) {
        const v16bf bh = *(const v16bf*)&fhi[rowoff + 32 * s];
        const v16bf bl = *(const v16bf*)&flo[rowoff + 32 * s];
        acc = __builtin_amdgcn_wmma_f32_16x16x32_bf16(
            false, Ahi[s], false, bh, (short)0, acc, false, false);
        acc = __builtin_amdgcn_wmma_f32_16x16x32_bf16(
            false, Alo[s], false, bh, (short)0, acc, false, false);
        acc = __builtin_amdgcn_wmma_f32_16x16x32_bf16(
            false, Ahi[s], false, bl, (short)0, acc, false, false);
      }
#pragma unroll
      for (int r = 0; r < 8; ++r)
        S[(r + 8 * half) * CHUNK + T * 16 + l16] = acc[r];
    }
    __syncthreads();                    // S complete

    // ---- Selection: threshold-filtered running top-16 ----
    if (t == 0) cnt[qrp] = 0;
    const float tq = tau[qrp];
    for (int j = 0; j < CHUNK / 16; ++j) {
      const int i = t + 16 * j;
      const float v = S[qrp * CHUNK + i];
      if (v > tq) {
        const int p = atomicAdd(&cnt[qrp], 1);
        if (p < CAP) { lv[qrp][16 + p] = v; li[qrp][16 + p] = gbase + i; }
      }
    }
    lv[qrp][t] = runval;
    li[qrp][t] = runidx;
    const int C = cnt[qrp];
    if (C > 0) {
      const bool big = C > CAP;
      const int  L   = 16 + (big ? 0 : C);
      for (int r = 0; r < KK; ++r) {
        float bv = NEGINF; int bp = 0x7fffffff;
        for (int p = t; p < L; p += 16) {
          const float v = lv[qrp][p];
          if (v > bv) { bv = v; bp = p; }
        }
        if (big) {
          for (int j = 0; j < CHUNK / 16; ++j) {
            const int i = t + 16 * j;
            const float v = S[qrp * CHUNK + i];
            if (v > bv) { bv = v; bp = LISTMAX + i; }
          }
        }
        for (int off = 8; off > 0; off >>= 1) {
          const float ov = __shfl_xor(bv, off, 16);
          const int   op = __shfl_xor(bp, off, 16);
          if (ov > bv || (ov == bv && op < bp)) { bv = ov; bp = op; }
        }
        if (t == 0) {
          nrv[qrp][r] = bv;
          if (bp < LISTMAX) { nri[qrp][r] = li[qrp][bp]; lv[qrp][bp] = NEGINF; }
          else { nri[qrp][r] = gbase + (bp - LISTMAX);
                 S[qrp * CHUNK + (bp - LISTMAX)] = NEGINF; }
          if (r == KK - 1) tau[qrp] = bv;
        }
      }
      runval = nrv[qrp][t];
      runidx = nri[qrp][t];
    }
  }

  const size_t e = ((size_t)(qt * 16 + qrp) * parts + part) * 16 + t;
  wsv[e] = runval;
  wsi[e] = runidx;
}

// ---------------------------------------------------------------------------
// Phase 1 (fallback, tiny ws): full-fp32 WMMA (16x16x4) version.
// ---------------------------------------------------------------------------
__global__ __launch_bounds__(256) void sim_topk_part_f32(
    const float* __restrict__ qry, const float* __restrict__ feat,
    float* __restrict__ wsv, int* __restrict__ wsi, int parts, int qpp)
{
  __shared__ float Aq[16 * DD];
  __shared__ float S[16 * CHUNK];
  __shared__ float lv[16][LISTMAX];
  __shared__ int   li[16][LISTMAX];
  __shared__ int   cnt[16];
  __shared__ float nrv[16][16];
  __shared__ int   nri[16][16];
  __shared__ float tau[16];

  const int tid  = threadIdx.x;
  const int lane = tid & 31;
  const int wave = tid >> 5;
  const int half = lane >> 4;
  const int l16  = lane & 15;
  const int qrp  = tid >> 4;
  const int t    = tid & 15;

  const int qt       = blockIdx.x;
  const int part     = blockIdx.y;
  const int partBase = part * qpp;
  const int nchunks  = qpp / CHUNK;
  const float NEGINF = -__builtin_inff();

  for (int i = tid; i < 16 * DD; i += 256) Aq[i] = qry[(size_t)qt * 16 * DD + i];
  if (t == 0) tau[qrp] = NEGINF;
  __syncthreads();

  v2f Areg[32];
#pragma unroll
  for (int s = 0; s < 32; ++s)
    Areg[s] = *(const v2f*)&Aq[l16 * DD + 4 * s + 2 * half];

  float runval = NEGINF;
  int   runidx = 0;

  for (int c = 0; c < nchunks; ++c) {
    __syncthreads();
    const int gbase = partBase + c * CHUNK;

#pragma unroll 1
    for (int j = 0; j < 4; ++j) {
      const int T = wave + 8 * j;
      const float* fb = feat + (size_t)(gbase + T * 16 + l16) * DD + 2 * half;
      v2f Breg[32];
#pragma unroll
      for (int s = 0; s < 32; ++s) Breg[s] = *(const v2f*)&fb[4 * s];
      v8f acc = {};
#pragma unroll
      for (int s = 0; s < 32; ++s)
        acc = __builtin_amdgcn_wmma_f32_16x16x4_f32(
            false, Areg[s], false, Breg[s], (short)0, acc, false, false);
#pragma unroll
      for (int r = 0; r < 8; ++r)
        S[(r + 8 * half) * CHUNK + T * 16 + l16] = acc[r];
    }
    __syncthreads();

    if (t == 0) cnt[qrp] = 0;
    const float tq = tau[qrp];
    for (int j = 0; j < CHUNK / 16; ++j) {
      const int i = t + 16 * j;
      const float v = S[qrp * CHUNK + i];
      if (v > tq) {
        const int p = atomicAdd(&cnt[qrp], 1);
        if (p < CAP) { lv[qrp][16 + p] = v; li[qrp][16 + p] = gbase + i; }
      }
    }
    lv[qrp][t] = runval;
    li[qrp][t] = runidx;
    const int C = cnt[qrp];
    if (C > 0) {
      const bool big = C > CAP;
      const int  L   = 16 + (big ? 0 : C);
      for (int r = 0; r < KK; ++r) {
        float bv = NEGINF; int bp = 0x7fffffff;
        for (int p = t; p < L; p += 16) {
          const float v = lv[qrp][p];
          if (v > bv) { bv = v; bp = p; }
        }
        if (big) {
          for (int j = 0; j < CHUNK / 16; ++j) {
            const int i = t + 16 * j;
            const float v = S[qrp * CHUNK + i];
            if (v > bv) { bv = v; bp = LISTMAX + i; }
          }
        }
        for (int off = 8; off > 0; off >>= 1) {
          const float ov = __shfl_xor(bv, off, 16);
          const int   op = __shfl_xor(bp, off, 16);
          if (ov > bv || (ov == bv && op < bp)) { bv = ov; bp = op; }
        }
        if (t == 0) {
          nrv[qrp][r] = bv;
          if (bp < LISTMAX) { nri[qrp][r] = li[qrp][bp]; lv[qrp][bp] = NEGINF; }
          else { nri[qrp][r] = gbase + (bp - LISTMAX);
                 S[qrp * CHUNK + (bp - LISTMAX)] = NEGINF; }
          if (r == KK - 1) tau[qrp] = bv;
        }
      }
      runval = nrv[qrp][t];
      runidx = nri[qrp][t];
    }
  }

  const size_t e = ((size_t)(qt * 16 + qrp) * parts + part) * 16 + t;
  wsv[e] = runval;
  wsi[e] = runidx;
}

// ---------------------------------------------------------------------------
// Phase 2: merge per-part candidates -> global top-16, then write all outputs.
// ---------------------------------------------------------------------------
__global__ __launch_bounds__(128) void merge_gather(
    const float* __restrict__ qry, const float* __restrict__ feat,
    const float* __restrict__ wsv, const int* __restrict__ wsi,
    float* __restrict__ out, int parts)
{
  __shared__ float cv[128];
  __shared__ int   ci[128];
  __shared__ int   sel[KK];

  const int tid = threadIdx.x;
  const int qi  = blockIdx.x;
  const int L   = parts * 16;
  const float NEGINF = -__builtin_inff();

  if (tid < L) {
    cv[tid] = wsv[(size_t)qi * L + tid];
    ci[tid] = wsi[(size_t)qi * L + tid];
  }
  __syncthreads();

  if (tid < 32) {
    for (int r = 0; r < KK; ++r) {
      float bv = NEGINF; int bp = 0x7fffffff;
      for (int p = tid; p < L; p += 32) {
        const float v = cv[p];
        if (v > bv) { bv = v; bp = p; }
      }
      for (int off = 16; off > 0; off >>= 1) {
        const float ov = __shfl_xor(bv, off, 32);
        const int   op = __shfl_xor(bp, off, 32);
        if (ov > bv || (ov == bv && op < bp)) { bv = ov; bp = op; }
      }
      if (tid == 0) { sel[r] = ci[bp]; cv[bp] = NEGINF; }
    }
  }
  __syncthreads();

  out[(size_t)qi * DD + tid] = qry[(size_t)qi * DD + tid];
  const size_t SELB = (size_t)BQ * DD;
#pragma unroll
  for (int j = 0; j < KK; ++j)
    out[SELB + ((size_t)qi * KK + j) * DD + tid] = feat[(size_t)sel[j] * DD + tid];
  const size_t EB = (size_t)(BQ + BQ * KK) * DD;
  if (tid < KK) {
    out[EB + (size_t)qi * KK + tid] = (float)qi;
    out[EB + (size_t)BQ * KK + (size_t)qi * KK + tid] = (float)(BQ + tid);
  }
}

// ---------------------------------------------------------------------------
extern "C" void kernel_launch(void* const* d_in, const int* in_sizes, int n_in,
                              void* d_out, int out_size, void* d_ws, size_t ws_size,
                              hipStream_t stream)
{
  (void)in_sizes; (void)n_in; (void)out_size;
  const float* qry  = (const float*)d_in[0];
  const float* feat = (const float*)d_in[1];
  float* out = (float*)d_out;
  char* ws = (char*)d_ws;

  const size_t FBANK = (size_t)QN * DD * 2;   // one bf16 feature bank (32 MB)
  const size_t QBANK = (size_t)BQ * DD * 2;   // one bf16 query bank (512 KB)
  const size_t CAND8 = (size_t)BQ * 8 * 16 * 8;
  const size_t needFast = 2 * FBANK + 2 * QBANK + CAND8;   // ~67 MB

  if (ws_size >= needFast) {
    // ---- fast path: pre-split bf16 hi/lo banks + 16x16x32 bf16 WMMA ----
    __bf16* fhi = (__bf16*)(ws);
    __bf16* flo = (__bf16*)(ws + FBANK);
    __bf16* qhi = (__bf16*)(ws + 2 * FBANK);
    __bf16* qlo = (__bf16*)(ws + 2 * FBANK + QBANK);
    float*  wsv = (float*)(ws + 2 * FBANK + 2 * QBANK);
    int*    wsi = (int*)((char*)wsv + (size_t)BQ * 8 * 16 * sizeof(float));

    split_bf16<<<(QN * DD) / 1024, 256, 0, stream>>>(feat, fhi, flo, QN * DD);
    split_bf16<<<(BQ * DD) / 1024, 256, 0, stream>>>(qry, qhi, qlo, BQ * DD);

    const int parts = 8, qpp = QN / parts;
    dim3 g1(BQ / 16, parts);
    sim_topk_part_bf16<<<g1, 256, 0, stream>>>(qhi, qlo, fhi, flo, wsv, wsi, parts, qpp);
    merge_gather<<<BQ, 128, 0, stream>>>(qry, feat, wsv, wsi, out, parts);
  } else {
    // ---- fallback: fp32 WMMA, tiny candidate workspace ----
    int parts = 8;
    if (ws_size < (size_t)BQ * 8 * 16 * 8)
      parts = (ws_size >= (size_t)BQ * 2 * 16 * 8) ? 2 : 1;
    const int qpp = QN / parts;
    float* wsv = (float*)ws;
    int*   wsi = (int*)(ws + (size_t)BQ * parts * 16 * sizeof(float));
    dim3 g1(BQ / 16, parts);
    sim_topk_part_f32<<<g1, 256, 0, stream>>>(qry, feat, wsv, wsi, parts, qpp);
    merge_gather<<<BQ, 128, 0, stream>>>(qry, feat, wsv, wsi, out, parts);
  }
}
